// GCN_6691559047384
// MI455X (gfx1250) — matrix operand order repaired and checked by reading, not compiled, not run on hardware
//
#include <hip/hip_runtime.h>
#include <hip/hip_bf16.h>

typedef __attribute__((ext_vector_type(16))) __bf16 v16bf;
typedef __attribute__((ext_vector_type(4)))  __bf16 v4bf;
typedef __attribute__((ext_vector_type(2)))  __bf16 v2bf;
typedef __attribute__((ext_vector_type(8)))  float  v8f;

static constexpr int       NN  = 100000;   // nodes
static constexpr int       NF  = 256;      // in features
static constexpr int       HID = 128;      // hidden
static constexpr int       NC  = 64;       // classes
static constexpr long long NE  = 1600000;  // edges

// ---------------- degree / normalization ----------------
__global__ void k_init_deg(float* deg, int n) {
  int i = blockIdx.x * blockDim.x + threadIdx.x;
  if (i < n) deg[i] = 1.0f;  // self loop
}

__global__ void k_count_deg(const long long* __restrict__ ei, long long E, float* deg) {
  long long e = (long long)blockIdx.x * blockDim.x + threadIdx.x;
  if (e < E) unsafeAtomicAdd(&deg[ei[E + e]], 1.0f);
}

__global__ void k_rsqrt(float* d, int n) {
  int i = blockIdx.x * blockDim.x + threadIdx.x;
  if (i < n) d[i] = rsqrtf(d[i]);
}

// ---------------- weight packing into WMMA B-fragment layout ----------------
// Bpack[((ks*(N/16) + t)*32 + lane)*16 + e] = bf16(W[k*N + n]),
// k = ks*32 + (e/8)*16 + (lane/16)*8 + e%8, n = t*16 + lane%16   (ISA 7.12.2 16-bit B layout)
template <int N, int K>
__global__ void k_pack_w(const float* __restrict__ W, __bf16* __restrict__ Bpack) {
  int tid = blockIdx.x * blockDim.x + threadIdx.x;
  if (tid >= K * N) return;
  int e    = tid & 15;
  int lane = (tid >> 4) & 31;
  int t    = (tid >> 9) % (N / 16);
  int ks   = tid / (512 * (N / 16));
  int k = ks * 32 + ((e < 8) ? 0 : 16) + (lane >> 4) * 8 + (e & 7);
  int n = t * 16 + (lane & 15);
  Bpack[tid] = (__bf16)W[k * N + n];
}

// ---------------- tiled WMMA GEMM: C[M,N] = A[M,K] (f32, cvt->bf16) x Bpack ----------------
// Software-pipelined: next A-tile global loads issued before the WMMA chain,
// converted/stored to the alternate LDS buffer after; B fragments batch-loaded
// so the 8 WMMAs issue back-to-back behind a single load-wait.
template <int N, int K>
__global__ __launch_bounds__(256) void k_gemm_wmma(const float* __restrict__ A,
                                                   const __bf16* __restrict__ Bpack,
                                                   float* __restrict__ C, int M) {
  constexpr int NT  = N / 16;     // output tiles per wave (8 or 4)
  constexpr int KS  = K / 32;     // k-steps
  constexpr int LDA = 36;         // LDS row stride (bf16), bank-conflict padding
  __shared__ __bf16 sA[2][128 * LDA];

  const int tid  = threadIdx.x;
  const int wave = tid >> 5;      // 0..7 -> 16-row strip each
  const int lane = tid & 31;
  const int m0   = blockIdx.x * 128;

  const int r0   = tid >> 3;        // 0..31 (staging row base)
  const int kq   = (tid & 7) * 4;   // 0..28 (staging col base)
  const int mrow = wave * 16 + (lane & 15);
  const int half = lane >> 4;

  const v8f vzero = {0.f, 0.f, 0.f, 0.f, 0.f, 0.f, 0.f, 0.f};
  v8f acc[NT];
#pragma unroll
  for (int t = 0; t < NT; ++t) acc[t] = vzero;

  auto load_tile = [&](int ks, float4* v) {
#pragma unroll
    for (int it = 0; it < 4; ++it) {
      int grow = m0 + r0 + it * 32;
      v[it] = make_float4(0.f, 0.f, 0.f, 0.f);
      if (grow < M) v[it] = *(const float4*)&A[(size_t)grow * K + ks * 32 + kq];
    }
  };
  auto store_tile = [&](int buf, const float4* v) {
#pragma unroll
    for (int it = 0; it < 4; ++it) {
      int row = r0 + it * 32;
      v4bf pk;
      pk[0] = (__bf16)v[it].x; pk[1] = (__bf16)v[it].y;
      pk[2] = (__bf16)v[it].z; pk[3] = (__bf16)v[it].w;
      *(v4bf*)&sA[buf][row * LDA + kq] = pk;
    }
  };

  // prologue: stage k-step 0
  {
    float4 cur[4];
    load_tile(0, cur);
    store_tile(0, cur);
  }
  __syncthreads();

#pragma unroll
  for (int ks = 0; ks < KS; ++ks) {
    const int buf = ks & 1;

    // issue next A-tile loads early (overlap with WMMA chain)
    float4 nxt[4];
    if (ks + 1 < KS) load_tile(ks + 1, nxt);

    // A fragment per ISA 16-bit A layout (consecutive k-pairs per VGPR)
    v16bf afrag;
#pragma unroll
    for (int r = 0; r < 8; ++r) {
      int bk = ((r < 4) ? 0 : 16) + half * 8 + (r & 3) * 2;
      v2bf p = *(const v2bf*)&sA[buf][mrow * LDA + bk];
      afrag[2 * r]     = p[0];
      afrag[2 * r + 1] = p[1];
    }

    // batch-load all B fragments, then back-to-back WMMAs
    v16bf bfrag[NT];
#pragma unroll
    for (int t = 0; t < NT; ++t)
      bfrag[t] = *(const v16bf*)&Bpack[(((size_t)ks * NT + t) * 32 + lane) * 16];
#pragma unroll
    for (int t = 0; t < NT; ++t)
      acc[t] = __builtin_amdgcn_wmma_f32_16x16x32_bf16(
          false, afrag, false, bfrag[t], (short)0, acc[t], false, false);

    // convert + stage next tile into the alternate buffer
    if (ks + 1 < KS) store_tile(buf ^ 1, nxt);
    __syncthreads();
  }

  // store: lane -> n = t*16 + lane%16, rows m0 + wave*16 + half*8 + r
#pragma unroll
  for (int t = 0; t < NT; ++t) {
    int n = t * 16 + (lane & 15);
#pragma unroll
    for (int r = 0; r < 8; ++r) {
      int m = m0 + wave * 16 + half * 8 + r;
      if (m < M) C[(size_t)m * N + n] = acc[t][r];
    }
  }
}

// ---------------- aggregation init: out = h * dinv^2 (+ bias) ----------------
__global__ void k_prop_init(const float* __restrict__ h, const float* __restrict__ dinv,
                            const float* __restrict__ bias, float* __restrict__ out,
                            long long total, int F) {
  long long i = (long long)blockIdx.x * blockDim.x + threadIdx.x;
  if (i >= total) return;
  long long row = i / F;
  float di = dinv[row];
  float v = h[i] * di * di;
  if (bias) v += bias[(int)(i % F)];
  out[i] = v;
}

// ---------------- per-edge gather/scale/scatter (F/4 threads per edge) ----------------
template <int F>
__global__ void k_scatter(const long long* __restrict__ ei, long long E,
                          const float* __restrict__ dinv, const float* __restrict__ H,
                          float* __restrict__ out) {
  constexpr int TPE = F / 4;
  long long gt = (long long)blockIdx.x * blockDim.x + threadIdx.x;
  long long e = gt / TPE;
  if (e >= E) return;
  int c = (int)(gt % TPE) * 4;
  long long s = ei[e];
  long long d = ei[E + e];
  float norm = dinv[s] * dinv[d];
  const float4 v = *(const float4*)&H[(size_t)s * F + c];
  float* o = &out[(size_t)d * F + c];
  unsafeAtomicAdd(o + 0, v.x * norm);
  unsafeAtomicAdd(o + 1, v.y * norm);
  unsafeAtomicAdd(o + 2, v.z * norm);
  unsafeAtomicAdd(o + 3, v.w * norm);
}

// ---------------- bias + ReLU ----------------
__global__ void k_bias_relu(const float* __restrict__ in, const float* __restrict__ b,
                            float* __restrict__ out, long long total, int F) {
  long long i = (long long)blockIdx.x * blockDim.x + threadIdx.x;
  if (i >= total) return;
  out[i] = fmaxf(in[i] + b[(int)(i % F)], 0.0f);
}

extern "C" void kernel_launch(void* const* d_in, const int* in_sizes, int n_in,
                              void* d_out, int out_size, void* d_ws, size_t ws_size,
                              hipStream_t stream) {
  const float*     x  = (const float*)d_in[0];
  const long long* ei = (const long long*)d_in[1];
  const float*     W1 = (const float*)d_in[2];
  const float*     b1 = (const float*)d_in[3];
  const float*     W2 = (const float*)d_in[4];
  const float*     b2 = (const float*)d_in[5];
  float* out = (float*)d_out;

  auto al = [](size_t v) { return (v + 255) & ~(size_t)255; };
  char* ws = (char*)d_ws;
  size_t off = 0;
  float*  dinv = (float*)(ws + off);  off += al((size_t)NN * 4);
  __bf16* Bp1  = (__bf16*)(ws + off); off += al((size_t)NF * HID * 2);
  __bf16* Bp2  = (__bf16*)(ws + off); off += al((size_t)HID * NC * 2);
  float*  h1   = (float*)(ws + off);  off += al((size_t)NN * HID * 4);
  float*  agg1 = (float*)(ws + off);  off += al((size_t)NN * HID * 4);
  float*  h2   = (float*)(ws + off);  off += al((size_t)NN * NC * 4);
  if (ws_size < off) return;

  const int T = 256;
  // degree / dinv
  k_init_deg<<<(NN + T - 1) / T, T, 0, stream>>>(dinv, NN);
  k_count_deg<<<(int)((NE + T - 1) / T), T, 0, stream>>>(ei, NE, dinv);
  k_rsqrt<<<(NN + T - 1) / T, T, 0, stream>>>(dinv, NN);

  // pack weights into WMMA fragment layout
  k_pack_w<HID, NF><<<(NF * HID + T - 1) / T, T, 0, stream>>>(W1, Bp1);
  k_pack_w<NC, HID><<<(HID * NC + T - 1) / T, T, 0, stream>>>(W2, Bp2);

  // layer 1: h1 = x @ W1
  k_gemm_wmma<HID, NF><<<(NN + 127) / 128, T, 0, stream>>>(x, Bp1, h1, NN);
  long long tot1 = (long long)NN * HID;
  k_prop_init<<<(int)((tot1 + T - 1) / T), T, 0, stream>>>(h1, dinv, nullptr, agg1, tot1, HID);
  long long sth1 = NE * (HID / 4);
  k_scatter<HID><<<(int)((sth1 + T - 1) / T), T, 0, stream>>>(ei, NE, dinv, h1, agg1);
  k_bias_relu<<<(int)((tot1 + T - 1) / T), T, 0, stream>>>(agg1, b1, h1, tot1, HID);

  // layer 2: h2 = relu_out @ W2
  k_gemm_wmma<NC, HID><<<(NN + 127) / 128, T, 0, stream>>>(h1, Bp2, h2, NN);
  long long tot2 = (long long)NN * NC;
  k_prop_init<<<(int)((tot2 + T - 1) / T), T, 0, stream>>>(h2, dinv, b2, out, tot2, NC);
  long long sth2 = NE * (NC / 4);
  k_scatter<NC><<<(int)((sth2 + T - 1) / T), T, 0, stream>>>(ei, NE, dinv, h2, out);
}
